// KoLeoLoss_34522947125368
// MI455X (gfx1250) — compile-verified, matrix operand order checked by reading, not run on hardware
//
#include <hip/hip_runtime.h>
#include <hip/hip_bf16.h>
#include <math.h>

#define N_ROWS 16384
#define D_HALF 384
#define D_FULL 768
#define EPSV   1e-8f

#define MB  128            // rows per block (4 M-waves x 32 rows)
#define NT  128            // cols per tile  (2 N-waves x 64 cols)
#define KC  128            // k per LDS chunk (4 wmma steps of 32)
#define LDA (KC + 8)       // padded LDS stride in bf16 (136 -> bank-conflict free, 16B aligned)

typedef __attribute__((ext_vector_type(16))) __bf16 v16bf;
typedef __attribute__((ext_vector_type(8)))  float  v8f;

// ---------------- Kernel 1: concat + L2 normalize (fp32 + bf16 copies) ----------------
__global__ __launch_bounds__(256) void normalize_kernel(const float* __restrict__ A,
                                                        const float* __restrict__ B,
                                                        float* __restrict__ Xf,
                                                        __bf16* __restrict__ Xb) {
  const int row = blockIdx.x;
  const int tid = threadIdx.x;
  __shared__ float red[256];
  float v[3];
  float ss = 0.f;
#pragma unroll
  for (int i = 0; i < 3; ++i) {
    int c = tid + i * 256;
    float x = (c < D_HALF) ? A[(size_t)row * D_HALF + c]
                           : B[(size_t)row * D_HALF + (c - D_HALF)];
    v[i] = x;
    ss += x * x;
  }
  red[tid] = ss;
  __syncthreads();
  for (int s = 128; s > 0; s >>= 1) {
    if (tid < s) red[tid] += red[tid + s];
    __syncthreads();
  }
  const float inv = 1.0f / fmaxf(sqrtf(red[0]), EPSV);
#pragma unroll
  for (int i = 0; i < 3; ++i) {
    int c = tid + i * 256;
    float x = v[i] * inv;
    Xf[(size_t)row * D_FULL + c] = x;
    Xb[(size_t)row * D_FULL + c] = (__bf16)x;
  }
}

// ---------------- Kernel 2: fused X*X^T (bf16 WMMA) + diag-masked row argmax ----------------
// Wave grid: 4 M-waves x 2 N-waves. Each wave: 32 rows (2 A-strips) x 64 cols (4 B-subtiles).
// B fragments reused by 2 WMMAs, A fragments by 4 -> 12 ds_load_b128 per 8 WMMAs.
__global__ __launch_bounds__(256) void dots_argmax_kernel(const __bf16* __restrict__ Xb,
                                                          int* __restrict__ Iout) {
  extern __shared__ char smem[];
  __bf16* As = (__bf16*)smem;               // MB x LDA
  __bf16* Bs = As + MB * LDA;               // NT x LDA

  const int tid  = threadIdx.x;
  const int lane = tid & 31;
  const int wave = tid >> 5;
  const int mw   = wave >> 1;               // 0..3 -> 32-row strip pair
  const int nw   = wave & 1;                // 0..1 -> 64-col half
  const int half = lane >> 4;               // C-matrix M split
  const int l16  = lane & 15;
  const int rowBlock = blockIdx.x * MB;
  const int kbase = half * 8;               // ISA 16-bit A/B fragment K split

  // per-lane running argmax: 2 strips x 8 rows
  float bestV[2][8];
  int   bestI[2][8];
#pragma unroll
  for (int s = 0; s < 2; ++s)
#pragma unroll
    for (int r = 0; r < 8; ++r) { bestV[s][r] = -3.0e38f; bestI[s][r] = 0; }

  for (int nTile = 0; nTile < N_ROWS; nTile += NT) {
    v8f acc[2][4];
#pragma unroll
    for (int s = 0; s < 2; ++s)
#pragma unroll
      for (int t = 0; t < 4; ++t) { v8f z = {}; acc[s][t] = z; }

    for (int kc = 0; kc < D_FULL; kc += KC) {
      __syncthreads();
      // cooperative load A chunk: rows rowBlock..+127, cols kc..kc+127
      for (int idx = tid; idx < MB * (KC / 8); idx += 256) {
        int r = idx >> 4;
        int c = idx & 15;
        *(uint4*)(As + r * LDA + c * 8) =
            *(const uint4*)(Xb + (size_t)(rowBlock + r) * D_FULL + kc + c * 8);
      }
      // cooperative load B chunk: rows nTile..+127 (dots columns), same k range
      for (int idx = tid; idx < NT * (KC / 8); idx += 256) {
        int r = idx >> 4;
        int c = idx & 15;
        *(uint4*)(Bs + r * LDA + c * 8) =
            *(const uint4*)(Xb + (size_t)(nTile + r) * D_FULL + kc + c * 8);
      }
      __syncthreads();

#pragma unroll
      for (int ks = 0; ks < KC; ks += 32) {
        union Frag { v16bf v; uint4 q[2]; };
        Frag af[2];
#pragma unroll
        for (int s = 0; s < 2; ++s) {
          const __bf16* ap = As + (mw * 32 + s * 16 + l16) * LDA + ks + kbase;
          af[s].q[0] = *(const uint4*)(ap);
          af[s].q[1] = *(const uint4*)(ap + 16);
        }
        Frag bfg[4];
#pragma unroll
        for (int t = 0; t < 4; ++t) {
          const __bf16* bp = Bs + (nw * 64 + t * 16 + l16) * LDA + ks + kbase;
          bfg[t].q[0] = *(const uint4*)(bp);
          bfg[t].q[1] = *(const uint4*)(bp + 16);
        }
        // keep all 6 fragments live; only non-memory instrs may cross
        __builtin_amdgcn_sched_barrier(1);
#pragma unroll
        for (int s = 0; s < 2; ++s)
#pragma unroll
          for (int t = 0; t < 4; ++t)
            acc[s][t] = __builtin_amdgcn_wmma_f32_16x16x32_bf16(
                false, af[s].v, false, bfg[t].v, (short)0, acc[s][t], false, false);
      }
    }

    // fold this 32x64 tile into the running per-lane argmax
#pragma unroll
    for (int s = 0; s < 2; ++s)
#pragma unroll
      for (int t = 0; t < 4; ++t) {
        const int col = nTile + nw * 64 + t * 16 + l16;
#pragma unroll
        for (int r = 0; r < 8; ++r) {
          const int rowg = rowBlock + mw * 32 + s * 16 + half * 8 + r;
          float vv = acc[s][t][r];
          if (col == rowg) vv = -1.0f;        // diagonal mask (reference sets -1)
          if (vv > bestV[s][r]) { bestV[s][r] = vv; bestI[s][r] = col; }
        }
      }
  }

  // cross-lane reduce across the 16 lanes that share each row (xor<16 stays in half)
#pragma unroll
  for (int s = 0; s < 2; ++s)
#pragma unroll
    for (int r = 0; r < 8; ++r) {
#pragma unroll
      for (int m = 1; m < 16; m <<= 1) {
        float ov = __shfl_xor(bestV[s][r], m, 32);
        int   oi = __shfl_xor(bestI[s][r], m, 32);
        if (ov > bestV[s][r] || (ov == bestV[s][r] && oi < bestI[s][r])) {
          bestV[s][r] = ov; bestI[s][r] = oi;
        }
      }
    }

  // cross-wave combine: the two N-waves each saw half the columns of every row
  __syncthreads();                            // done with As/Bs, reuse LDS
  float* sV = (float*)smem;                   // [2][128]
  int*   sI = (int*)(smem + 2 * MB * sizeof(float));
  if (l16 == 0) {
#pragma unroll
    for (int s = 0; s < 2; ++s)
#pragma unroll
      for (int r = 0; r < 8; ++r) {
        int rl = mw * 32 + s * 16 + half * 8 + r;
        sV[nw * MB + rl] = bestV[s][r];
        sI[nw * MB + rl] = bestI[s][r];
      }
  }
  __syncthreads();
  if (tid < MB) {
    float v0 = sV[tid], v1 = sV[MB + tid];
    int   i0 = sI[tid], i1 = sI[MB + tid];
    int best = (v1 > v0 || (v1 == v0 && i1 < i0)) ? i1 : i0;
    Iout[rowBlock + tid] = best;
  }
}

// ---------------- Kernel 3: per-row log(||x_i - x_I[i] + eps|| + eps) ----------------
__global__ __launch_bounds__(256) void dist_log_kernel(const float* __restrict__ Xf,
                                                       const int* __restrict__ I,
                                                       float* __restrict__ logd) {
  const int row = blockIdx.x;
  const int tid = threadIdx.x;
  const int j = I[row];
  __shared__ float red[256];
  float ss = 0.f;
#pragma unroll
  for (int i = 0; i < 3; ++i) {
    int c = tid + i * 256;
    float d = Xf[(size_t)row * D_FULL + c] - Xf[(size_t)j * D_FULL + c] + EPSV;
    ss += d * d;
  }
  red[tid] = ss;
  __syncthreads();
  for (int s = 128; s > 0; s >>= 1) {
    if (tid < s) red[tid] += red[tid + s];
    __syncthreads();
  }
  if (tid == 0) logd[row] = logf(sqrtf(red[0]) + EPSV);
}

// ---------------- Kernel 4: loss = -mean(logd) ----------------
__global__ __launch_bounds__(256) void reduce_loss_kernel(const float* __restrict__ logd,
                                                          float* __restrict__ out) {
  const int tid = threadIdx.x;
  __shared__ float red[256];
  float s = 0.f;
  for (int i = tid; i < N_ROWS; i += 256) s += logd[i];
  red[tid] = s;
  __syncthreads();
  for (int st = 128; st > 0; st >>= 1) {
    if (tid < st) red[tid] += red[tid + st];
    __syncthreads();
  }
  if (tid == 0) out[0] = -red[0] / (float)N_ROWS;
}

extern "C" void kernel_launch(void* const* d_in, const int* in_sizes, int n_in,
                              void* d_out, int out_size, void* d_ws, size_t ws_size,
                              hipStream_t stream) {
  const float* a = (const float*)d_in[0];
  const float* b = (const float*)d_in[1];
  float* out = (float*)d_out;

  char* ws = (char*)d_ws;
  float*  Xf   = (float*)ws;                                          // 48 MB
  __bf16* Xb   = (__bf16*)(ws + (size_t)N_ROWS * D_FULL * 4);         // 24 MB
  int*    I    = (int*)(ws + (size_t)N_ROWS * D_FULL * 4
                           + (size_t)N_ROWS * D_FULL * 2);            // 64 KB
  float*  logd = (float*)((char*)I + (size_t)N_ROWS * 4);             // 64 KB

  normalize_kernel<<<N_ROWS, 256, 0, stream>>>(a, b, Xf, Xb);

  const size_t smem = (size_t)2 * MB * LDA * sizeof(__bf16);          // ~68 KB dynamic LDS
  dots_argmax_kernel<<<N_ROWS / MB, 256, smem, stream>>>(Xb, I);

  dist_log_kernel<<<N_ROWS, 256, 0, stream>>>(Xf, I, logd);
  reduce_loss_kernel<<<1, 256, 0, stream>>>(logd, out);
}
